// CdwExtractor_29051158790655
// MI455X (gfx1250) — compile-verified, matrix operand order checked by reading, not run on hardware
//
#include <hip/hip_runtime.h>
#include <math.h>

#define BATCH 8
#define NPTS  4096
#define KNN   16

typedef __attribute__((ext_vector_type(16))) _Float16 v16h;
typedef __attribute__((ext_vector_type(8)))  _Float16 v8h;
typedef __attribute__((ext_vector_type(8)))  float    v8f;
typedef __attribute__((ext_vector_type(4)))  int      v4i;

#if defined(__has_builtin)
#if __has_builtin(__builtin_amdgcn_global_load_async_to_lds_b128)
#define HAVE_ASYNC_LDS 1
#endif
#if __has_builtin(__builtin_amdgcn_s_wait_asynccnt)
#define HAVE_WAIT_ASYNC_BUILTIN 1
#endif
#endif

#if defined(HAVE_ASYNC_LDS)
typedef __attribute__((address_space(1))) v4i gas_v4i;
typedef __attribute__((address_space(3))) v4i las_v4i;
#endif

__device__ inline void wait_asynccnt0() {
#if defined(HAVE_WAIT_ASYNC_BUILTIN)
  __builtin_amdgcn_s_wait_asynccnt(0);
#else
  asm volatile("s_wait_asynccnt 0x0" ::: "memory");
#endif
}

// ---------------------------------------------------------------------------
// KNN + relative-feature construction (f16 output, stride 32, zero-padded)
// ---------------------------------------------------------------------------
__global__ __launch_bounds__(256) void knn_feat_kernel(
    const float* __restrict__ pts, _Float16* __restrict__ feat) {
  int b = blockIdx.y;
  int n = blockIdx.x * blockDim.x + threadIdx.x;
  const float* pb = pts + (size_t)b * NPTS * 3;
  float px = pb[(size_t)n * 3 + 0];
  float py = pb[(size_t)n * 3 + 1];
  float pz = pb[(size_t)n * 3 + 2];

  float bd[KNN + 1];
  int   bi[KNN + 1];
#pragma unroll
  for (int i = 0; i < KNN + 1; ++i) { bd[i] = 3.0e38f; bi[i] = 0; }

  __shared__ float sp[256][3];
  for (int m0 = 0; m0 < NPTS; m0 += 256) {
    __syncthreads();
    int t = threadIdx.x;
    sp[t][0] = pb[(size_t)(m0 + t) * 3 + 0];
    sp[t][1] = pb[(size_t)(m0 + t) * 3 + 1];
    sp[t][2] = pb[(size_t)(m0 + t) * 3 + 2];
    __syncthreads();
    for (int j = 0; j < 256; ++j) {
      float dx = px - sp[j][0];
      float dy = py - sp[j][1];
      float dz = pz - sp[j][2];
      float d  = dx * dx + dy * dy + dz * dz;
      if (d < bd[KNN]) {
        int m = m0 + j;
        int p = KNN;
        while (p > 0 && bd[p - 1] > d) {
          bd[p] = bd[p - 1]; bi[p] = bi[p - 1]; --p;
        }
        bd[p] = d; bi[p] = m;
      }
    }
  }

  int c0 = bi[0];
  float cx = pb[(size_t)c0 * 3 + 0];
  float cy = pb[(size_t)c0 * 3 + 1];
  float cz = pb[(size_t)c0 * 3 + 2];
  size_t base = ((size_t)(b * NPTS + n)) * KNN * 32;
#pragma unroll 4
  for (int k = 0; k < KNN; ++k) {
    int mm = bi[k + 1];
    float rx = pb[(size_t)mm * 3 + 0] - cx;
    float ry = pb[(size_t)mm * 3 + 1] - cy;
    float rz = pb[(size_t)mm * 3 + 2] - cz;
    float dist = sqrtf(rx * rx + ry * ry + rz * rz + 1e-8f);
    _Float16* f = feat + base + (size_t)k * 32;
    f[0] = (_Float16)cx; f[1] = (_Float16)cy; f[2] = (_Float16)cz;
    f[3] = (_Float16)rx; f[4] = (_Float16)ry; f[5] = (_Float16)rz;
    f[6] = (_Float16)dist;
#pragma unroll
    for (int c = 7; c < 32; ++c) f[c] = (_Float16)0.0f;
  }
}

// ---------------------------------------------------------------------------
// Pad-copy f32 [R,C] -> f16 [Rp,Cp] zero-filled (weights / pts)
// ---------------------------------------------------------------------------
__global__ __launch_bounds__(256) void pad_copy_h_kernel(
    const float* __restrict__ src, int R, int C,
    _Float16* __restrict__ dst, int Rp, int Cp) {
  size_t i = (size_t)blockIdx.x * 256 + threadIdx.x;
  if (i >= (size_t)Rp * Cp) return;
  int r = (int)(i / Cp), c = (int)(i % Cp);
  dst[i] = (r < R && c < C) ? (_Float16)src[(size_t)r * C + c] : (_Float16)0.0f;
}

// ---------------------------------------------------------------------------
// GEMM: Y[.,ldY](f32) = X[Mp,Kp](f16) * W[Op,Kp](f16)^T  — no load guards.
// Block tile 64x128, 8 waves, wave tile 16x64 -> 4 WMMA / K-step.
// Tiles staged via GLOBAL_LOAD_ASYNC_TO_LDS_B128 when available.
// ---------------------------------------------------------------------------
__device__ inline v16h frag16(const _Float16* rowp, int hf) {
  v8h lo = *(const v8h*)(rowp + hf);
  v8h hi = *(const v8h*)(rowp + 16 + hf);
  return __builtin_shufflevector(lo, hi, 0, 1, 2, 3, 4, 5, 6, 7,
                                 8, 9, 10, 11, 12, 13, 14, 15);
}

__global__ __launch_bounds__(256) void gemm_wmma_kernel(
    const _Float16* __restrict__ X, const _Float16* __restrict__ W,
    float* __restrict__ Y, int Kp, int ldY) {
  __shared__ _Float16 sX[64][40];
  __shared__ _Float16 sW[128][40];

  const int tid  = threadIdx.x;
  const int lane = tid & 31;
  const int wid  = tid >> 5;
  const int wr   = (wid & 3) * 16;   // wave row offset (M)
  const int wo   = (wid >> 2) * 64;  // wave col offset (O)

  v8f acc0 = {}, acc1 = {}, acc2 = {}, acc3 = {};

  const int trow = tid >> 2;          // 0..63
  const int tcol = (tid & 3) * 8;     // 0,8,16,24
  const _Float16* xp  = X + (size_t)(blockIdx.y * 64 + trow) * Kp + tcol;
  const _Float16* wp0 = W + (size_t)(blockIdx.x * 128 + trow) * Kp + tcol;
  const _Float16* wp1 = wp0 + (size_t)64 * Kp;
  _Float16* dX  = &sX[trow][tcol];
  _Float16* dW0 = &sW[trow][tcol];
  _Float16* dW1 = &sW[trow + 64][tcol];

  for (int k0 = 0; k0 < Kp; k0 += 32) {
    __syncthreads();
#if defined(HAVE_ASYNC_LDS)
    __builtin_amdgcn_global_load_async_to_lds_b128(
        (gas_v4i*)(xp + k0), (las_v4i*)dX, 0, 0);
    __builtin_amdgcn_global_load_async_to_lds_b128(
        (gas_v4i*)(wp0 + k0), (las_v4i*)dW0, 0, 0);
    __builtin_amdgcn_global_load_async_to_lds_b128(
        (gas_v4i*)(wp1 + k0), (las_v4i*)dW1, 0, 0);
    wait_asynccnt0();
#else
    *(v8h*)dX  = *(const v8h*)(xp + k0);
    *(v8h*)dW0 = *(const v8h*)(wp0 + k0);
    *(v8h*)dW1 = *(const v8h*)(wp1 + k0);
#endif
    __syncthreads();

    const int hf = (lane >> 4) * 8;
    const int rl = lane & 15;
    v16h a  = frag16(&sX[wr + rl][0], hf);
    v16h b0 = frag16(&sW[wo + rl][0], hf);
    v16h b1 = frag16(&sW[wo + 16 + rl][0], hf);
    v16h b2 = frag16(&sW[wo + 32 + rl][0], hf);
    v16h b3 = frag16(&sW[wo + 48 + rl][0], hf);
    acc0 = __builtin_amdgcn_wmma_f32_16x16x32_f16(false, a, false, b0,
                                                  (short)0, acc0, false, false);
    acc1 = __builtin_amdgcn_wmma_f32_16x16x32_f16(false, a, false, b1,
                                                  (short)0, acc1, false, false);
    acc2 = __builtin_amdgcn_wmma_f32_16x16x32_f16(false, a, false, b2,
                                                  (short)0, acc2, false, false);
    acc3 = __builtin_amdgcn_wmma_f32_16x16x32_f16(false, a, false, b3,
                                                  (short)0, acc3, false, false);
  }

  // Store. Guards are wave-uniform: each fragment spans 16 aligned columns
  // and ldY is a multiple of 16.
  const int nn    = lane & 15;
  const int mhi   = (lane >> 4) * 8;
  const int cbase = blockIdx.x * 128 + wo;
  float* yb = Y + (size_t)(blockIdx.y * 64 + wr + mhi) * ldY + nn;
  if (cbase < ldY) {
#pragma unroll
    for (int v = 0; v < 8; ++v) yb[(size_t)v * ldY + cbase] = acc0[v];
  }
  if (cbase + 16 < ldY) {
#pragma unroll
    for (int v = 0; v < 8; ++v) yb[(size_t)v * ldY + cbase + 16] = acc1[v];
  }
  if (cbase + 32 < ldY) {
#pragma unroll
    for (int v = 0; v < 8; ++v) yb[(size_t)v * ldY + cbase + 32] = acc2[v];
  }
  if (cbase + 48 < ldY) {
#pragma unroll
    for (int v = 0; v < 8; ++v) yb[(size_t)v * ldY + cbase + 48] = acc3[v];
  }
}

// ---------------------------------------------------------------------------
// BatchNorm: stats (one block per channel, f32 input) + apply (f32 -> f16)
// ---------------------------------------------------------------------------
__global__ __launch_bounds__(256) void bn_stats_kernel(
    const float* __restrict__ Y, float* __restrict__ stats, int M, int O, int ld) {
  int o = blockIdx.x;
  float s = 0.0f, s2 = 0.0f;
  for (int m = threadIdx.x; m < M; m += 256) {
    float v = Y[(size_t)m * ld + o];
    s += v; s2 += v * v;
  }
  __shared__ float rs[256];
  __shared__ float rq[256];
  rs[threadIdx.x] = s; rq[threadIdx.x] = s2;
  __syncthreads();
  for (int st = 128; st > 0; st >>= 1) {
    if (threadIdx.x < st) {
      rs[threadIdx.x] += rs[threadIdx.x + st];
      rq[threadIdx.x] += rq[threadIdx.x + st];
    }
    __syncthreads();
  }
  if (threadIdx.x == 0) { stats[o] = rs[0]; stats[O + o] = rq[0]; }
}

__global__ __launch_bounds__(256) void bn_apply_kernel(
    const float* __restrict__ Y, const float* __restrict__ stats,
    const float* __restrict__ g, const float* __restrict__ b,
    _Float16* __restrict__ out, int M, int O, int ld, int Mp, int relu) {
  size_t i = (size_t)blockIdx.x * 256 + threadIdx.x;
  if (i >= (size_t)Mp * ld) return;
  int o = (int)(i % ld);
  if (o >= O) { out[i] = (_Float16)0.0f; return; }  // keep zero padding
  float invM = 1.0f / (float)M;
  float mean = stats[o] * invM;
  float var  = stats[O + o] * invM - mean * mean;
  float v = (Y[i] - mean) * rsqrtf(var + 1e-5f) * g[o] + b[o];
  if (relu) v = fmaxf(v, 0.0f);
  out[i] = (_Float16)v;
}

// ---------------------------------------------------------------------------
// Reductions / concats / elementwise (f16 activations)
// ---------------------------------------------------------------------------
__global__ __launch_bounds__(256) void maxk_kernel(
    const _Float16* __restrict__ h, _Float16* __restrict__ out, int P, int C) {
  size_t i = (size_t)blockIdx.x * 256 + threadIdx.x;
  if (i >= (size_t)P * C) return;
  int p = (int)(i / C), c = (int)(i % C);
  float m = -3.0e38f;
#pragma unroll
  for (int k = 0; k < KNN; ++k)
    m = fmaxf(m, (float)h[((size_t)p * KNN + k) * 32 + c]);
  out[i] = (_Float16)m;
}

__global__ __launch_bounds__(256) void maxn_kernel(
    const _Float16* __restrict__ x, _Float16* __restrict__ out,
    int N, int C, int ld) {
  int b = blockIdx.x / C;
  int c = blockIdx.x % C;
  float m = -3.0e38f;
  for (int n = threadIdx.x; n < N; n += 256)
    m = fmaxf(m, (float)x[((size_t)b * N + n) * ld + c]);
  __shared__ float rm[256];
  rm[threadIdx.x] = m;
  __syncthreads();
  for (int st = 128; st > 0; st >>= 1) {
    if (threadIdx.x < st) rm[threadIdx.x] = fmaxf(rm[threadIdx.x], rm[threadIdx.x + st]);
    __syncthreads();
  }
  if (threadIdx.x == 0) out[(size_t)b * C + c] = (_Float16)rm[0];
}

__global__ __launch_bounds__(256) void concat2_kernel(
    const _Float16* __restrict__ A, int ca, const _Float16* __restrict__ B, int cb,
    _Float16* __restrict__ out, int M) {
  int ct = ca + cb;
  size_t i = (size_t)blockIdx.x * 256 + threadIdx.x;
  if (i >= (size_t)M * ct) return;
  int r = (int)(i / ct), c = (int)(i % ct);
  out[i] = (c < ca) ? A[(size_t)r * ca + c] : B[(size_t)r * cb + (c - ca)];
}

__global__ __launch_bounds__(256) void concat_bcast_kernel(
    const _Float16* __restrict__ A, int ca, const _Float16* __restrict__ G, int cg,
    _Float16* __restrict__ out, int M, int N) {
  int ct = ca + cg;
  size_t i = (size_t)blockIdx.x * 256 + threadIdx.x;
  if (i >= (size_t)M * ct) return;
  int r = (int)(i / ct), c = (int)(i % ct);
  out[i] = (c < ca) ? A[(size_t)r * ca + c]
                    : G[(size_t)(r / N) * cg + (c - ca)];
}

__global__ __launch_bounds__(256) void concat4_kernel(
    const _Float16* __restrict__ f1, const _Float16* __restrict__ f2,
    const _Float16* __restrict__ f3, const _Float16* __restrict__ g3,
    _Float16* __restrict__ out, int M, int N) {
  size_t i = (size_t)blockIdx.x * 256 + threadIdx.x;
  if (i >= (size_t)M * 352) return;
  int r = (int)(i / 352), c = (int)(i % 352);
  _Float16 v;
  if (c < 32)       v = f1[(size_t)r * 32 + c];
  else if (c < 96)  v = f2[(size_t)r * 64 + (c - 32)];
  else if (c < 224) v = f3[(size_t)r * 128 + (c - 96)];
  else              v = g3[(size_t)(r / N) * 128 + (c - 224)];
  out[i] = v;
}

__global__ __launch_bounds__(256) void add_relu_kernel(
    const _Float16* __restrict__ a, const _Float16* __restrict__ b,
    _Float16* __restrict__ out, size_t n) {
  size_t i = (size_t)blockIdx.x * 256 + threadIdx.x;
  if (i >= n) return;
  out[i] = (_Float16)fmaxf((float)a[i] + (float)b[i], 0.0f);
}

__global__ __launch_bounds__(256) void copy_kernel(
    const float* __restrict__ src, float* __restrict__ dst, size_t n) {
  size_t i = (size_t)blockIdx.x * 256 + threadIdx.x;
  if (i >= n) return;
  dst[i] = src[i];
}

// ---------------------------------------------------------------------------
// Fused softmax over N + attention pooling (one block per (b,o))
// ---------------------------------------------------------------------------
__global__ __launch_bounds__(256) void softmax_att_kernel(
    const float* __restrict__ logits, const _Float16* __restrict__ ftr4,
    _Float16* __restrict__ att, int N, int C) {
  int b = blockIdx.x / C;
  int o = blockIdx.x % C;
  const float*    L = logits + (size_t)b * N * C + o;
  const _Float16* F = ftr4   + (size_t)b * N * C + o;

  __shared__ float r1[256];
  __shared__ float r2[256];

  float m = -3.0e38f;
  for (int n = threadIdx.x; n < N; n += 256)
    m = fmaxf(m, L[(size_t)n * C]);
  r1[threadIdx.x] = m;
  __syncthreads();
  for (int st = 128; st > 0; st >>= 1) {
    if (threadIdx.x < st) r1[threadIdx.x] = fmaxf(r1[threadIdx.x], r1[threadIdx.x + st]);
    __syncthreads();
  }
  float mx = r1[0];
  __syncthreads();

  float s = 0.0f, ws = 0.0f;
  for (int n = threadIdx.x; n < N; n += 256) {
    float e = expf(L[(size_t)n * C] - mx);
    s  += e;
    ws += e * (float)F[(size_t)n * C];
  }
  r1[threadIdx.x] = s; r2[threadIdx.x] = ws;
  __syncthreads();
  for (int st = 128; st > 0; st >>= 1) {
    if (threadIdx.x < st) {
      r1[threadIdx.x] += r1[threadIdx.x + st];
      r2[threadIdx.x] += r2[threadIdx.x + st];
    }
    __syncthreads();
  }
  if (threadIdx.x == 0) att[(size_t)b * C + o] = (_Float16)(r2[0] / r1[0]);
}

// ---------------------------------------------------------------------------
// Host side
// ---------------------------------------------------------------------------
static inline void gemm(const _Float16* X, const _Float16* W, float* Y,
                        int Mp, int Kp, int Op, int ldY, hipStream_t s) {
  dim3 grid(Op / 128, Mp / 64);
  gemm_wmma_kernel<<<grid, 256, 0, s>>>(X, W, Y, Kp, ldY);
}

static inline void bn(const float* Y, float* stats, const float* g,
                      const float* b, _Float16* out, int M, int O, int ld,
                      int Mp, int relu, hipStream_t s) {
  bn_stats_kernel<<<O, 256, 0, s>>>(Y, stats, M, O, ld);
  size_t total = (size_t)Mp * ld;
  bn_apply_kernel<<<(unsigned)((total + 255) / 256), 256, 0, s>>>(
      Y, stats, g, b, out, M, O, ld, Mp, relu);
}

static inline void padw(const float* src, int O, int K, _Float16* dst,
                        int Op, int Kp, hipStream_t s) {
  size_t total = (size_t)Op * Kp;
  pad_copy_h_kernel<<<(unsigned)((total + 255) / 256), 256, 0, s>>>(
      src, O, K, dst, Op, Kp);
}

extern "C" void kernel_launch(void* const* d_in, const int* in_sizes, int n_in,
                              void* d_out, int out_size, void* d_ws, size_t ws_size,
                              hipStream_t stream) {
  (void)in_sizes; (void)n_in; (void)out_size; (void)ws_size;

  const float* pts    = (const float*)d_in[0];
  const float* ws_f[39];
  for (int i = 0; i < 39; ++i) ws_f[i] = (const float*)d_in[i];

  const float* a1_g = ws_f[2],  *a1_b = ws_f[3];
  const float* a2_g = ws_f[5],  *a2_b = ws_f[6];
  const float* lift_g = ws_f[8], *lift_b = ws_f[9];
  const float* mix_g = ws_f[11], *mix_b = ws_f[12];
  const float* r1a_g = ws_f[14], *r1a_b = ws_f[15];
  const float* r1b_g = ws_f[17], *r1b_b = ws_f[18];
  const float* r1s_g = ws_f[20], *r1s_b = ws_f[21];
  const float* r2a_g = ws_f[23], *r2a_b = ws_f[24];
  const float* r2b_g = ws_f[26], *r2b_b = ws_f[27];
  const float* fuse_g = ws_f[29], *fuse_b = ws_f[30];
  const float* fc1_g = ws_f[33], *fc1_b = ws_f[34];
  const float* fc2_g = ws_f[36], *fc2_b = ws_f[37];

  const int P  = BATCH * NPTS;        // 32768
  const int M1 = P * KNN;             // 524288

  char* base = (char*)d_ws;
  size_t off = 0;
  auto allocb = [&](size_t bytes) {
    void* p = base + off;
    off += (bytes + 255) & ~(size_t)255;
    return p;
  };
  auto ah = [&](size_t n) { return (_Float16*)allocb(n * 2); };
  auto af = [&](size_t n) { return (float*)allocb(n * 4); };

  // f16 padded weights
  _Float16* a1p   = ah(128 * 32);
  _Float16* a2p   = ah(128 * 32);
  _Float16* liftp = ah(128 * 32);
  _Float16* mixp  = ah(128 * 64);
  _Float16* r1ap  = ah(128 * 32);
  _Float16* r1bp  = ah(128 * 32);
  _Float16* r1sp  = ah(128 * 32);
  _Float16* r2ap  = ah(128 * 128);
  _Float16* r2bp  = ah(128 * 128);
  _Float16* fusep = ah(512 * 352);
  _Float16* attp  = ah(512 * 512);
  _Float16* fc1p  = ah(512 * 1024);
  _Float16* fc2p  = ah(1024 * 512);
  _Float16* fc3p  = ah(1024 * 1024);

  // f16 activations
  _Float16* ptsp  = ah((size_t)P * 32);
  _Float16* feat  = ah((size_t)M1 * 32);
  _Float16* h1a   = ah((size_t)M1 * 32);
  _Float16* h2a   = ah((size_t)M1 * 32);
  _Float16* nbr   = ah((size_t)P * 32);
  _Float16* lifty = ah((size_t)P * 32);
  _Float16* cat64 = ah((size_t)P * 64);
  _Float16* ftr1  = ah((size_t)P * 32);
  _Float16* r1aa  = ah((size_t)P * 32);
  _Float16* t1a   = ah((size_t)P * 64);
  _Float16* sc1a  = ah((size_t)P * 64);
  _Float16* ftr2  = ah((size_t)P * 64);
  _Float16* g2    = ah(512);
  _Float16* x3    = ah((size_t)P * 128);
  _Float16* r2aa  = ah((size_t)P * 128);
  _Float16* t2a   = ah((size_t)P * 128);
  _Float16* ftr3  = ah((size_t)P * 128);
  _Float16* g3    = ah(1024);
  _Float16* ftr4  = ah((size_t)P * 512);
  _Float16* f4max = ah(4096);
  _Float16* attv  = ah(4096);
  _Float16* hcat  = ah((size_t)64 * 1024);
  _Float16* fc1a  = ah((size_t)64 * 512);
  _Float16* fc2a  = ah((size_t)64 * 1024);

  // f32 scratch: shared pre-BN GEMM output (max M1*32 == P*512 == 16.8M)
  float* gy    = af((size_t)M1 * 32);
  float* fc3y  = af((size_t)64 * 1024);
  float* stats = af(2048);
  // cat352 aliases dead early region (feat; 11.5M halves fits in 16.8M)
  _Float16* cat352 = feat;
  float*    logits = gy;   // att logits live in gy (read by softmax)

  // 0. pad/convert weights + pts
  padw(ws_f[1], 16, 7, a1p, 128, 32, stream);
  padw(ws_f[4], 32, 16, a2p, 128, 32, stream);
  padw(ws_f[7], 32, 3, liftp, 128, 32, stream);
  padw(ws_f[10], 32, 64, mixp, 128, 64, stream);
  padw(ws_f[13], 32, 32, r1ap, 128, 32, stream);
  padw(ws_f[16], 64, 32, r1bp, 128, 32, stream);
  padw(ws_f[19], 64, 32, r1sp, 128, 32, stream);
  padw(ws_f[22], 128, 128, r2ap, 128, 128, stream);
  padw(ws_f[25], 128, 128, r2bp, 128, 128, stream);
  padw(ws_f[28], 512, 352, fusep, 512, 352, stream);
  padw(ws_f[31], 512, 512, attp, 512, 512, stream);
  padw(ws_f[32], 512, 1024, fc1p, 512, 1024, stream);
  padw(ws_f[35], 1024, 512, fc2p, 1024, 512, stream);
  padw(ws_f[38], 1024, 1024, fc3p, 1024, 1024, stream);
  padw(pts, P, 3, ptsp, P, 32, stream);

  // 1. KNN features
  knn_feat_kernel<<<dim3(NPTS / 256, BATCH), 256, 0, stream>>>(pts, feat);

  // 2-3. a1 (7->16), a2 (16->32)
  gemm(feat, a1p, gy, M1, 32, 128, 32, stream);
  bn(gy, stats, a1_g, a1_b, h1a, M1, 16, 32, M1, 1, stream);
  gemm(h1a, a2p, gy, M1, 32, 128, 32, stream);
  bn(gy, stats, a2_g, a2_b, h2a, M1, 32, 32, M1, 1, stream);

  // 4. neighbor max-pool
  maxk_kernel<<<(unsigned)(((size_t)P * 32 + 255) / 256), 256, 0, stream>>>(h2a, nbr, P, 32);

  // 5. lift (3->32)
  gemm(ptsp, liftp, gy, P, 32, 128, 32, stream);
  bn(gy, stats, lift_g, lift_b, lifty, P, 32, 32, P, 1, stream);

  // 6-7. mix (64->32)
  concat2_kernel<<<(unsigned)(((size_t)P * 64 + 255) / 256), 256, 0, stream>>>(
      lifty, 32, nbr, 32, cat64, P);
  gemm(cat64, mixp, gy, P, 64, 128, 32, stream);
  bn(gy, stats, mix_g, mix_b, ftr1, P, 32, 32, P, 1, stream);

  // 8-11. residual block 1
  gemm(ftr1, r1ap, gy, P, 32, 128, 32, stream);
  bn(gy, stats, r1a_g, r1a_b, r1aa, P, 32, 32, P, 1, stream);
  gemm(r1aa, r1bp, gy, P, 32, 128, 64, stream);
  bn(gy, stats, r1b_g, r1b_b, t1a, P, 64, 64, P, 0, stream);
  gemm(ftr1, r1sp, gy, P, 32, 128, 64, stream);
  bn(gy, stats, r1s_g, r1s_b, sc1a, P, 64, 64, P, 0, stream);
  add_relu_kernel<<<(unsigned)(((size_t)P * 64 + 255) / 256), 256, 0, stream>>>(
      sc1a, t1a, ftr2, (size_t)P * 64);

  // 12-16. residual block 2 with global context
  maxn_kernel<<<BATCH * 64, 256, 0, stream>>>(ftr2, g2, NPTS, 64, 64);
  concat_bcast_kernel<<<(unsigned)(((size_t)P * 128 + 255) / 256), 256, 0, stream>>>(
      ftr2, 64, g2, 64, x3, P, NPTS);
  gemm(x3, r2ap, gy, P, 128, 128, 128, stream);
  bn(gy, stats, r2a_g, r2a_b, r2aa, P, 128, 128, P, 1, stream);
  gemm(r2aa, r2bp, gy, P, 128, 128, 128, stream);
  bn(gy, stats, r2b_g, r2b_b, t2a, P, 128, 128, P, 0, stream);
  add_relu_kernel<<<(unsigned)(((size_t)P * 128 + 255) / 256), 256, 0, stream>>>(
      x3, t2a, ftr3, (size_t)P * 128);

  // 17-19. fuse (352->512)
  maxn_kernel<<<BATCH * 128, 256, 0, stream>>>(ftr3, g3, NPTS, 128, 128);
  concat4_kernel<<<(unsigned)(((size_t)P * 352 + 255) / 256), 256, 0, stream>>>(
      ftr1, ftr2, ftr3, g3, cat352, P, NPTS);
  gemm(cat352, fusep, gy, P, 352, 512, 512, stream);
  bn(gy, stats, fuse_g, fuse_b, ftr4, P, 512, 512, P, 1, stream);

  // 20-22. attention pooling + global max
  gemm(ftr4, attp, logits, P, 512, 512, 512, stream);
  softmax_att_kernel<<<BATCH * 512, 256, 0, stream>>>(logits, ftr4, attv, NPTS, 512);
  maxn_kernel<<<BATCH * 512, 256, 0, stream>>>(ftr4, f4max, NPTS, 512, 512);

  // 23-26. FC head (M=8 padded to 64 rows)
  concat2_kernel<<<(unsigned)(((size_t)BATCH * 1024 + 255) / 256), 256, 0, stream>>>(
      f4max, 512, attv, 512, hcat, BATCH);
  gemm(hcat, fc1p, gy, 64, 1024, 512, 512, stream);
  bn(gy, stats, fc1_g, fc1_b, fc1a, BATCH, 512, 512, 64, 1, stream);
  gemm(fc1a, fc2p, gy, 64, 512, 1024, 1024, stream);
  bn(gy, stats, fc2_g, fc2_b, fc2a, BATCH, 1024, 1024, 64, 1, stream);
  gemm(fc2a, fc3p, fc3y, 64, 1024, 1024, 1024, stream);
  copy_kernel<<<(unsigned)(((size_t)BATCH * 1024 + 255) / 256), 256, 0, stream>>>(
      fc3y, (float*)d_out, (size_t)BATCH * 1024);
}
